// GPT_17643725652396
// MI455X (gfx1250) — matrix operand order repaired and checked by reading, not compiled, not run on hardware
//
#include <hip/hip_runtime.h>
#include <math.h>

typedef __attribute__((ext_vector_type(16))) __bf16 v16bf;
typedef __attribute__((ext_vector_type(8)))  float  v8f;
typedef __attribute__((ext_vector_type(4)))  unsigned int u32x4;

#define BB   4
#define TT   1024
#define CC   768
#define LL   12
#define HH   12
#define VV   50257
#define HSZ  64
#define ROWS (BB*TT)     /* 4096 */
#define VPAD 50304       /* V padded to multiple of 128 for tiling (393*128) */

// ---------- helpers ----------
__device__ __forceinline__ unsigned short f2bf(float f) {
  unsigned int x = __float_as_uint(f);
  unsigned int r = x + 0x7FFFu + ((x >> 16) & 1u);   // round-to-nearest-even
  return (unsigned short)(r >> 16);
}

// Low 32 bits of a generic pointer to __shared__ == LDS byte offset (aperture mapping).
__device__ __forceinline__ unsigned lds_off32(const void* p) {
  return (unsigned)(unsigned long long)p;
}

// CDNA5 async global->LDS copy (ASYNCcnt-tracked, GV addressing mode).
__device__ __forceinline__ void async_g2l_b128(unsigned lds_off, const void* gaddr) {
  asm volatile("global_load_async_to_lds_b128 %0, %1, off"
               :: "v"(lds_off), "v"(gaddr) : "memory");
}
__device__ __forceinline__ void wait_async0() {
  asm volatile("s_wait_asynccnt 0" ::: "memory");
}

union FragU { unsigned int u[8]; u32x4 q[2]; v16bf v; };

// ---------- embedding: x = wte[idx] + wpe[:T] ----------
__global__ void embed_kernel(const int* __restrict__ idx, const float* __restrict__ wte,
                             const float* __restrict__ wpe, float* __restrict__ x) {
  int i = blockIdx.x * blockDim.x + threadIdx.x;
  if (i >= ROWS * CC) return;
  int c   = i % CC;
  int row = i / CC;
  int t   = row % TT;
  int id  = idx[row];
  x[i] = wte[(size_t)id * CC + c] + wpe[(size_t)t * CC + c];
}

// ---------- layernorm: f32 row -> bf16 row ----------
__global__ void layernorm_kernel(const float* __restrict__ x, const float* __restrict__ w,
                                 const float* __restrict__ b, unsigned short* __restrict__ out) {
  int row = blockIdx.x;
  int tid = threadIdx.x;
  const float* xr = x + (size_t)row * CC;
  __shared__ float red[256];
  float s = 0.f;
  for (int i = tid; i < CC; i += 256) s += xr[i];
  red[tid] = s; __syncthreads();
  for (int st = 128; st > 0; st >>= 1) { if (tid < st) red[tid] += red[tid + st]; __syncthreads(); }
  float mu = red[0] * (1.f / CC);
  __syncthreads();
  float v = 0.f;
  for (int i = tid; i < CC; i += 256) { float d = xr[i] - mu; v += d * d; }
  red[tid] = v; __syncthreads();
  for (int st = 128; st > 0; st >>= 1) { if (tid < st) red[tid] += red[tid + st]; __syncthreads(); }
  float rstd = rsqrtf(red[0] * (1.f / CC) + 1e-5f);
  unsigned short* orow = out + (size_t)row * CC;
  for (int i = tid; i < CC; i += 256)
    orow[i] = f2bf((xr[i] - mu) * rstd * w[i] + b[i]);
}

// ---------- pack f32 weight [K,N] (or transposed [N,K]) -> bf16 WMMA-B tiles ----------
// dst dword index = tile*256 + g*128 + lane*4 + e   (tile = kt*NT + nt, p = g*4 + e)
// dword holds bf16 pair for (K = kt*32 + (lane>=16?16:0) + 2p {+0,+1}, N = nt*16 + lane%16)
__global__ void pack_w_kernel(const float* __restrict__ src, unsigned int* __restrict__ dst,
                              int K, int Nsrc, int Npad, int transposed) {
  int idx = blockIdx.x * blockDim.x + threadIdx.x;
  int NT = Npad >> 4;
  int total = (K >> 5) * NT * 256;
  if (idx >= total) return;
  int e    = idx & 3;
  int lane = (idx >> 2) & 31;
  int g    = (idx >> 7) & 1;
  int tile = idx >> 8;
  int nt   = tile % NT;
  int kt   = tile / NT;
  int p    = g * 4 + e;
  int col  = (nt << 4) + (lane & 15);
  int k0   = (kt << 5) + ((lane >> 4) << 4) + (p << 1);
  float f0 = 0.f, f1 = 0.f;
  if (col < Nsrc) {
    if (transposed) {                 // src is [Nsrc, K] (wte): W[k][n] = src[n*K + k]
      f0 = src[(size_t)col * K + k0];
      f1 = src[(size_t)col * K + k0 + 1];
    } else {                          // src is [K, Nsrc]
      f0 = src[(size_t)k0 * Nsrc + col];
      f1 = src[(size_t)(k0 + 1) * Nsrc + col];
    }
  }
  dst[idx] = ((unsigned int)f2bf(f1) << 16) | (unsigned int)f2bf(f0);
}

// ---------- WMMA bf16 GEMM with async-LDS double buffering ----------
// Block: 256 thr = 8 waves (wm 0..1, wn 0..3). Wave tile 64x32, block tile 128x128.
// epi 0: f32 out (+bias);  epi 1: f32 out = resid + bias + gemm;  epi 2: bf16 out = gelu(bias+gemm)
#define A_STRIDE 80                    /* 64B row + 16B pad to spread LDS banks */
#define A_BUF    (128 * A_STRIDE)      /* 10240 B */
#define B_BUF    (8 * 1024)            /* 8 n-tiles * 1KB */

__global__ void __launch_bounds__(256)
gemm_wmma_kernel(const unsigned short* __restrict__ A, int lda,
                 const unsigned int* __restrict__ Bp, int NT,
                 const float* __restrict__ bias,
                 float* __restrict__ outf, unsigned short* __restrict__ outh,
                 const float* __restrict__ resid,
                 int M, int N, int K, int epi) {
  __shared__ __align__(16) char smemA[2 * A_BUF];
  __shared__ __align__(16) char smemB[2 * B_BUF];

  int tid  = threadIdx.x;
  int lane = tid & 31;
  int wave = tid >> 5;
  int wm = wave >> 2;                  // 0..1
  int wn = wave & 3;                   // 0..3
  int m0 = blockIdx.y * 128;
  int nt0 = blockIdx.x * 8;            // first n-tile of this block

  // ---- async copy issue helper (per thread: 2x A b128, 2x B b128) ----
  int arow  = tid >> 1;                // 0..127
  int ahalf = tid & 1;                 // 0..1 : 32B half of a 64B row
  int bj    = tid >> 5;                // 0..7 : n-tile within block
  const unsigned short* Abase = A + (size_t)(m0 + arow) * lda + ahalf * 16;
  const unsigned int*   Bbase = Bp + ((size_t)(nt0 + bj) << 8) + lane * 4;
  unsigned aoff = arow * A_STRIDE + ahalf * 32;
  unsigned boff = bj * 1024 + lane * 16;

  int KT = K >> 5;
#define ISSUE_COPIES(ktc, buf)                                                        \
  {                                                                                   \
    const unsigned short* ag = Abase + ((ktc) << 5);                                  \
    unsigned al = lds_off32(smemA) + (buf) * A_BUF + aoff;                            \
    async_g2l_b128(al,      ag);                                                      \
    async_g2l_b128(al + 16, ag + 8);                                                  \
    const unsigned int* bg = Bbase + (((size_t)(ktc) * NT) << 8);                     \
    unsigned bl = lds_off32(smemB) + (buf) * B_BUF + boff;                            \
    async_g2l_b128(bl,       bg);                                                     \
    async_g2l_b128(bl + 512, bg + 128);                                               \
  }

  ISSUE_COPIES(0, 0);
  wait_async0();
  __syncthreads();

  v8f acc[4][2] = {};
  int r  = lane & 15;
  int kb = ((lane >> 4) << 5);         // byte offset within A row: 0 or 32

  int buf = 0;
  for (int kt = 0; kt < KT; ++kt) {
    if (kt + 1 < KT) ISSUE_COPIES(kt + 1, buf ^ 1);

    const char* Ab = smemA + buf * A_BUF;
    const char* Bb = smemB + buf * B_BUF;

    FragU b[2];
#pragma unroll
    for (int j = 0; j < 2; ++j) {
      const char* bp = Bb + (wn * 2 + j) * 1024 + lane * 16;
      b[j].q[0] = *(const u32x4*)(bp);
      b[j].q[1] = *(const u32x4*)(bp + 512);
    }
#pragma unroll
    for (int i = 0; i < 4; ++i) {
      FragU a;
      const char* ap = Ab + (wm * 64 + i * 16 + r) * A_STRIDE + kb;
      a.q[0] = *(const u32x4*)(ap);
      a.q[1] = *(const u32x4*)(ap + 16);
#pragma unroll
      for (int j = 0; j < 2; ++j)
        acc[i][j] = __builtin_amdgcn_wmma_f32_16x16x32_bf16(
            false, a.v, false, b[j].v, (short)0, acc[i][j], false, false);
    }

    wait_async0();
    __syncthreads();
    buf ^= 1;
  }

#pragma unroll
  for (int i = 0; i < 4; ++i) {
#pragma unroll
    for (int j = 0; j < 2; ++j) {
      int tn = blockIdx.x * 128 + wn * 32 + j * 16 + (lane & 15);
      if (tn >= N) continue;
      float bv = bias ? bias[tn] : 0.f;
#pragma unroll
      for (int rr = 0; rr < 8; ++rr) {
        int tm = m0 + wm * 64 + i * 16 + rr + ((lane >> 4) << 3);
        float val = acc[i][j][rr] + bv;
        size_t o = (size_t)tm * N + tn;
        if (epi == 0) {
          outf[o] = val;
        } else if (epi == 1) {
          outf[o] = resid[o] + val;
        } else {
          float c = val;
          float g = 0.5f * c * (1.f + tanhf(0.7978845608028654f * (c + 0.044715f * c * c * c)));
          outh[o] = f2bf(g);
        }
      }
    }
  }
}

// ---------- split qkv (f32, [row][3C]) into head-major bf16 Q/K and transposed V ----------
// Qb,Kb: [B*H][T][64]   Vt: [B*H][64][T]
__global__ void qkv_split_kernel(const float* __restrict__ qkv,
                                 unsigned short* __restrict__ Qb,
                                 unsigned short* __restrict__ Kb,
                                 unsigned short* __restrict__ Vt) {
  int i = blockIdx.x * blockDim.x + threadIdx.x;
  if (i >= ROWS * CC) return;
  int c   = i % CC;
  int row = i / CC;
  int h = c / HSZ, d = c % HSZ;
  int b = row / TT, t = row % TT;
  int bh = b * HH + h;
  const float* src = qkv + (size_t)row * 3 * CC + c;
  Qb[((size_t)bh * TT + t) * HSZ + d] = f2bf(src[0]);
  Kb[((size_t)bh * TT + t) * HSZ + d] = f2bf(src[CC]);
  Vt[((size_t)bh * HSZ + d) * TT + t] = f2bf(src[2 * CC]);
}

// ---------- WMMA flash attention: one wave per 16-query tile ----------
// S = Q(16x64) @ K^T (2 wmma per 16-key subtile), online softmax, P transposed via
// wave-private LDS bounce, O += P(16x32) @ V (4 wmma per 32-key chunk).
#define P_STRIDE 36   /* ushorts per P row (72B) to spread LDS banks */
__global__ void __launch_bounds__(256)
attn_wmma_kernel(const unsigned short* __restrict__ Qb, const unsigned short* __restrict__ Kb,
                 const unsigned short* __restrict__ Vt, unsigned short* __restrict__ y) {
  __shared__ __align__(8) unsigned short pLds[8][16 * P_STRIDE];
  int wave = threadIdx.x >> 5, lane = threadIdx.x & 31;
  int wid = blockIdx.x * 8 + wave;          // 0 .. B*H*(T/16)-1 = 3071
  int qt = wid & 63;                        // query tile within sequence (T/16 = 64)
  int bh = wid >> 6;                        // 0..47
  int b  = bh / HH, h = bh % HH;
  const unsigned short* Qp = Qb + (size_t)bh * TT * HSZ;
  const unsigned short* Kp = Kb + (size_t)bh * TT * HSZ;
  const unsigned short* Vp = Vt + (size_t)bh * HSZ * TT;
  int ln = lane & 15;
  int hi = lane >> 4;

  // Q fragments (A operand), rows qt*16+ln, head-dim 0..31 / 32..63
  FragU aq[2];
  {
    const unsigned short* qrow = Qp + (size_t)(qt * 16 + ln) * HSZ;
#pragma unroll
    for (int s = 0; s < 2; ++s)
#pragma unroll
      for (int p = 0; p < 8; ++p) {
        int k = s * 32 + ((p >> 2) << 4) + ((p & 3) << 1) + hi * 8;
        aq[s].u[p] = *(const unsigned int*)(qrow + k);
      }
  }

  float mrow[8], lrow[8], corr[8];
#pragma unroll
  for (int rr = 0; rr < 8; ++rr) { mrow[rr] = -__builtin_inff(); lrow[rr] = 0.f; }
  v8f o[4] = {};
  int qhi = qt * 16 + 15;
  unsigned short* pbase = &pLds[wave][0];

  for (int j0 = 0; j0 <= qhi; j0 += 32) {
    // ---- S = scale * Q@K^T for keys j0..j0+31, causal-masked ----
    v8f S[2];
#pragma unroll
    for (int st = 0; st < 2; ++st) {
      const unsigned short* krow = Kp + (size_t)(j0 + st * 16 + ln) * HSZ;
      v8f s = {};
#pragma unroll
      for (int sd = 0; sd < 2; ++sd) {
        FragU bk;
#pragma unroll
        for (int p = 0; p < 8; ++p)
          bk.u[p] = *(const unsigned int*)(krow + sd * 32 + ((p >> 2) << 4) + ((p & 3) << 1) + hi * 8);
        s = __builtin_amdgcn_wmma_f32_16x16x32_bf16(false, aq[sd].v, false, bk.v,
                                                    (short)0, s, false, false);
      }
      int kabs = j0 + st * 16 + ln;         // this lane's key column
#pragma unroll
      for (int rr = 0; rr < 8; ++rr) {
        int qabs = qt * 16 + rr + hi * 8;
        float sv = s[rr] * 0.125f;          // 1/sqrt(64)
        s[rr] = (kabs <= qabs) ? sv : -__builtin_inff();
      }
      S[st] = s;
    }
    // ---- online softmax stats (rows span 16 lanes within each half) ----
#pragma unroll
    for (int rr = 0; rr < 8; ++rr) {
      float rm = fmaxf(S[0][rr], S[1][rr]);
#pragma unroll
      for (int off = 8; off > 0; off >>= 1) rm = fmaxf(rm, __shfl_xor(rm, off));
      float mn = fmaxf(mrow[rr], rm);
      corr[rr] = __expf(mrow[rr] - mn);
      mrow[rr] = mn;
    }
#pragma unroll
    for (int st = 0; st < 2; ++st)
#pragma unroll
      for (int rr = 0; rr < 8; ++rr)
        S[st][rr] = __expf(S[st][rr] - mrow[rr]);
#pragma unroll
    for (int rr = 0; rr < 8; ++rr) {
      float rs = S[0][rr] + S[1][rr];
#pragma unroll
      for (int off = 8; off > 0; off >>= 1) rs += __shfl_xor(rs, off);
      lrow[rr] = lrow[rr] * corr[rr] + rs;
    }
#pragma unroll
    for (int dt = 0; dt < 4; ++dt)
#pragma unroll
      for (int rr = 0; rr < 8; ++rr)
        o[dt][rr] *= corr[rr];
    // ---- transpose P (C/D layout -> A layout) through wave-private LDS ----
#pragma unroll
    for (int st = 0; st < 2; ++st)
#pragma unroll
      for (int rr = 0; rr < 8; ++rr)
        pbase[(rr + hi * 8) * P_STRIDE + st * 16 + ln] = f2bf(S[st][rr]);
    FragU ap;
#pragma unroll
    for (int p = 0; p < 8; ++p) {
      int k = hi * 16 + 2 * p;
      ap.u[p] = *(const unsigned int*)(pbase + ln * P_STRIDE + k);
    }
    // ---- O += P @ V  (V transposed: contiguous key pairs) ----
#pragma unroll
    for (int dt = 0; dt < 4; ++dt) {
      FragU bv;
      const unsigned short* vrow = Vp + (size_t)(dt * 16 + ln) * TT + j0 + hi * 16;
#pragma unroll
      for (int p = 0; p < 8; ++p)
        bv.u[p] = *(const unsigned int*)(vrow + 2 * p);
      o[dt] = __builtin_amdgcn_wmma_f32_16x16x32_bf16(false, ap.v, false, bv.v,
                                                      (short)0, o[dt], false, false);
    }
  }

  // ---- epilogue: y[b*T + q][h*64 + d] = O / l ----
#pragma unroll
  for (int dt = 0; dt < 4; ++dt)
#pragma unroll
    for (int rr = 0; rr < 8; ++rr) {
      int m = rr + hi * 8;
      float val = o[dt][rr] / lrow[rr];
      y[((size_t)(b * TT + qt * 16 + m)) * CC + h * HSZ + dt * 16 + ln] = f2bf(val);
    }
}

// ---------- loss ----------
__global__ void loss_init_kernel(float* loss) {
  if (blockIdx.x == 0 && threadIdx.x == 0) loss[0] = 0.f;
}

__global__ void loss_kernel(const float* __restrict__ logits, const int* __restrict__ targets,
                            float* __restrict__ loss) {
  int row = blockIdx.x;
  int tid = threadIdx.x;
  const float* lr = logits + (size_t)row * VV;
  __shared__ float red[256];
  float m = -__builtin_inff();
  for (int i = tid; i < VV; i += 256) m = fmaxf(m, lr[i]);
  red[tid] = m; __syncthreads();
  for (int st = 128; st > 0; st >>= 1) { if (tid < st) red[tid] = fmaxf(red[tid], red[tid + st]); __syncthreads(); }
  m = red[0]; __syncthreads();
  float s = 0.f;
  for (int i = tid; i < VV; i += 256) s += __expf(lr[i] - m);
  red[tid] = s; __syncthreads();
  for (int st = 128; st > 0; st >>= 1) { if (tid < st) red[tid] += red[tid + st]; __syncthreads(); }
  if (tid == 0) {
    float lse = m + logf(red[0]);
    float nll = lse - lr[targets[row]];
    atomicAdd(loss, nll * (1.f / ROWS));
  }
}

// ---------- host ----------
extern "C" void kernel_launch(void* const* d_in, const int* in_sizes, int n_in,
                              void* d_out, int out_size, void* d_ws, size_t ws_size,
                              hipStream_t stream) {
  (void)in_sizes; (void)n_in; (void)out_size; (void)ws_size;
  const int*   idx  = (const int*)d_in[0];
  const int*   tgt  = (const int*)d_in[1];
  const float* wte  = (const float*)d_in[2];
  const float* wpe  = (const float*)d_in[3];
  const float* ln1w = (const float*)d_in[4];
  const float* ln1b = (const float*)d_in[5];
  const float* qkvw = (const float*)d_in[6];
  const float* qkvb = (const float*)d_in[7];
  const float* apw  = (const float*)d_in[8];
  const float* apb  = (const float*)d_in[9];
  const float* ln2w = (const float*)d_in[10];
  const float* ln2b = (const float*)d_in[11];
  const float* fcw  = (const float*)d_in[12];
  const float* fcb  = (const float*)d_in[13];
  const float* mpw  = (const float*)d_in[14];
  const float* mpb  = (const float*)d_in[15];
  const float* lnfw = (const float*)d_in[16];
  const float* lnfb = (const float*)d_in[17];

  char* ws = (char*)d_ws;
  float*          x      = (float*)ws;          ws += (size_t)ROWS * CC * 4;
  unsigned short* h      = (unsigned short*)ws; ws += (size_t)ROWS * CC * 2;
  float*          qkv    = (float*)ws;          ws += (size_t)ROWS * 3 * CC * 4;
  unsigned short* y      = (unsigned short*)ws; ws += (size_t)ROWS * CC * 2;
  unsigned short* g      = (unsigned short*)ws; ws += (size_t)ROWS * 4 * CC * 2;
  unsigned short* Qb     = (unsigned short*)ws; ws += (size_t)ROWS * CC * 2;
  unsigned short* Kb     = (unsigned short*)ws; ws += (size_t)ROWS * CC * 2;
  unsigned short* Vt     = (unsigned short*)ws; ws += (size_t)ROWS * CC * 2;
  unsigned int*   wp_qkv = (unsigned int*)ws;   ws += (size_t)CC * 3 * CC * 2;
  unsigned int*   wp_ap  = (unsigned int*)ws;   ws += (size_t)CC * CC * 2;
  unsigned int*   wp_fc  = (unsigned int*)ws;   ws += (size_t)CC * 4 * CC * 2;
  unsigned int*   wp_mp  = (unsigned int*)ws;   ws += (size_t)4 * CC * CC * 2;
  unsigned int*   wp_lm  = (unsigned int*)ws;   ws += (size_t)CC * VPAD * 2;

  float* logits = (float*)d_out;
  float* loss   = logits + (size_t)ROWS * VV;

  embed_kernel<<<(ROWS * CC + 255) / 256, 256, 0, stream>>>(idx, wte, wpe, x);
  loss_init_kernel<<<1, 32, 0, stream>>>(loss);

  dim3 gqkv((3 * CC) / 128, ROWS / 128);
  dim3 gproj(CC / 128, ROWS / 128);
  dim3 gfc((4 * CC) / 128, ROWS / 128);

  for (int l = 0; l < LL; ++l) {
    layernorm_kernel<<<ROWS, 256, 0, stream>>>(x, ln1w + l * CC, ln1b + l * CC, h);

    int tq = (CC / 32) * (3 * CC / 16) * 256;
    pack_w_kernel<<<(tq + 255) / 256, 256, 0, stream>>>(qkvw + (size_t)l * CC * 3 * CC,
                                                        wp_qkv, CC, 3 * CC, 3 * CC, 0);
    gemm_wmma_kernel<<<gqkv, 256, 0, stream>>>(h, CC, wp_qkv, (3 * CC) / 16, qkvb + l * 3 * CC,
                                               qkv, nullptr, nullptr, ROWS, 3 * CC, CC, 0);

    qkv_split_kernel<<<(ROWS * CC + 255) / 256, 256, 0, stream>>>(qkv, Qb, Kb, Vt);
    attn_wmma_kernel<<<(BB * HH * (TT / 16)) / 8, 256, 0, stream>>>(Qb, Kb, Vt, y);

    int ta = (CC / 32) * (CC / 16) * 256;
    pack_w_kernel<<<(ta + 255) / 256, 256, 0, stream>>>(apw + (size_t)l * CC * CC,
                                                        wp_ap, CC, CC, CC, 0);
    gemm_wmma_kernel<<<gproj, 256, 0, stream>>>(y, CC, wp_ap, CC / 16, apb + l * CC,
                                                x, nullptr, x, ROWS, CC, CC, 1);

    layernorm_kernel<<<ROWS, 256, 0, stream>>>(x, ln2w + l * CC, ln2b + l * CC, h);

    int tf = (CC / 32) * (4 * CC / 16) * 256;
    pack_w_kernel<<<(tf + 255) / 256, 256, 0, stream>>>(fcw + (size_t)l * CC * 4 * CC,
                                                        wp_fc, CC, 4 * CC, 4 * CC, 0);
    gemm_wmma_kernel<<<gfc, 256, 0, stream>>>(h, CC, wp_fc, (4 * CC) / 16, fcb + l * 4 * CC,
                                              nullptr, g, nullptr, ROWS, 4 * CC, CC, 2);

    int tm = (4 * CC / 32) * (CC / 16) * 256;
    pack_w_kernel<<<(tm + 255) / 256, 256, 0, stream>>>(mpw + (size_t)l * 4 * CC * CC,
                                                        wp_mp, 4 * CC, CC, CC, 0);
    gemm_wmma_kernel<<<gproj, 256, 0, stream>>>(g, 4 * CC, wp_mp, CC / 16, mpb + l * CC,
                                                x, nullptr, x, ROWS, CC, 4 * CC, 1);
  }

  layernorm_kernel<<<ROWS, 256, 0, stream>>>(x, lnfw, lnfb, h);

  int tl = (CC / 32) * (VPAD / 16) * 256;
  pack_w_kernel<<<(tl + 255) / 256, 256, 0, stream>>>(wte, wp_lm, CC, VV, VPAD, 1);

  dim3 glm(VPAD / 128, ROWS / 128);
  gemm_wmma_kernel<<<glm, 256, 0, stream>>>(h, CC, wp_lm, VPAD / 16, nullptr,
                                            logits, nullptr, nullptr, ROWS, VV, CC, 0);

  loss_kernel<<<ROWS, 256, 0, stream>>>(logits, tgt, loss);
}
